// PraxisMemory_57449482551795
// MI455X (gfx1250) — matrix operand order repaired and checked by reading, not compiled, not run on hardware
//
#include <hip/hip_runtime.h>
#include <hip/hip_bf16.h>

typedef __attribute__((ext_vector_type(16))) __bf16 v16bf;
typedef __attribute__((ext_vector_type(8)))  __bf16 v8bf;
typedef __attribute__((ext_vector_type(8)))  float  v8f;

#define Hh 16
#define Qn 4096   // B*S per head
#define Dd 128
#define Mm 4096
#define TQ 128    // queries per block (4 waves x 32 queries)
#define NC 64     // memory rows per LDS chunk
#define PAD 8     // bf16 elements of row padding in LDS

#define QT_BYTES (TQ * (Dd + PAD) * 2)   // 34816
#define KT_BYTES (NC * (Dd + PAD) * 2)   // 17408 (>= 128*16*8 topk bytes)

// ---------- helpers: 64-bit shuffles within 16-lane groups (wave32) ----------
__device__ __forceinline__ unsigned long long shfl_xor_u64_w16(unsigned long long v, int m) {
    int lo = __shfl_xor((int)(unsigned)(v & 0xffffffffull), m, 16);
    int hi = __shfl_xor((int)(unsigned)(v >> 32), m, 16);
    return ((unsigned long long)(unsigned)hi << 32) | (unsigned)lo;
}
__device__ __forceinline__ unsigned long long shfl_get_u64_w16(unsigned long long v, int src) {
    int lo = __shfl((int)(unsigned)(v & 0xffffffffull), src, 16);
    int hi = __shfl((int)(unsigned)(v >> 32), src, 16);
    return ((unsigned long long)(unsigned)hi << 32) | (unsigned)lo;
}

// sortable key: larger float -> larger u64; idx in low 32 bits
__device__ __forceinline__ unsigned long long packKey(float s, unsigned idx) {
    unsigned u = __float_as_uint(s);
    u = (u & 0x80000000u) ? ~u : (u | 0x80000000u);
    return ((unsigned long long)u << 32) | (unsigned long long)idx;
}
__device__ __forceinline__ float unpackScore(unsigned long long k) {
    unsigned u = (unsigned)(k >> 32);
    u = (u & 0x80000000u) ? (u ^ 0x80000000u) : ~u;
    return __uint_as_float(u);
}

// full bitonic sort (ascending) of 16 values held one-per-lane in a 16-lane group
__device__ __forceinline__ unsigned long long sort16_asc(unsigned long long v, int g) {
#pragma unroll
    for (int k = 2; k <= 16; k <<= 1) {
#pragma unroll
        for (int j = k >> 1; j >= 1; j >>= 1) {
            unsigned long long p = shfl_xor_u64_w16(v, j);
            bool up      = ((g & k) == 0);
            bool upper   = ((g & j) != 0);
            bool takeMax = (upper == up);
            unsigned long long mx = (v > p) ? v : p;
            unsigned long long mn = (v > p) ? p : v;
            v = takeMax ? mx : mn;
        }
    }
    return v;
}
// bitonic merge of a bitonic sequence -> descending sorted
__device__ __forceinline__ unsigned long long merge16_desc(unsigned long long v, int g) {
#pragma unroll
    for (int j = 8; j >= 1; j >>= 1) {
        unsigned long long p = shfl_xor_u64_w16(v, j);
        bool takeMax = ((g & j) == 0);
        unsigned long long mx = (v > p) ? v : p;
        unsigned long long mn = (v > p) ? p : v;
        v = takeMax ? mx : mn;
    }
    return v;
}

// streaming top-16 merge of one 16x16 score tile into the per-wave sorted lists.
// Hot path is ONE v_cmp_gt_f32 + uniform branch per v; all u64 work is on the
// rarely-taken merge path. topMin[v] caches the current list minimum
// (broadcast to all lanes) so no DPP/shuffle is needed in the hot path.
__device__ __forceinline__ void topk_update(const v8f& c, unsigned long long* topP,
                                            float* topMin, unsigned idxBase, int g) {
#pragma unroll
    for (int v = 0; v < 8; ++v) {
        if (__ballot(c[v] > topMin[v])) {          // wave-uniform threshold filter
            unsigned long long cand = packKey(c[v], idxBase + (unsigned)g);
            cand = sort16_asc(cand, g);            // ascending
            unsigned long long m = (topP[v] > cand) ? topP[v] : cand;  // bitonic
            m = merge16_desc(m, g);                // back to descending sorted
            topP[v] = m;
            topMin[v] = unpackScore(shfl_get_u64_w16(m, 15));  // refresh threshold
        }
    }
}

// ---------------- kernel 1: L2-normalize rows -> bf16 workspace ----------------
// rows [0,65536): qn  (query transposed [B,H,S,D] -> [H,B*S,D])
// rows [65536,131072): kn (key_memories [H,M,D])
__launch_bounds__(256)
__global__ void praxis_normalize_kernel(const float* __restrict__ query,
                                        const float* __restrict__ keyMem,
                                        __bf16* __restrict__ qnW,
                                        __bf16* __restrict__ knW)
{
    const int row  = blockIdx.x * 8 + (threadIdx.x >> 5);
    const int lane = threadIdx.x & 31;

    const float* src;
    __bf16* dst;
    if (row < Hh * Qn) {
        const int h  = row >> 12;         // / 4096
        const int qi = row & 4095;
        const int b  = qi >> 11;          // / 2048
        const int s  = qi & 2047;
        src = query + ((((size_t)b * Hh + h) * 2048 + s) * Dd);
        dst = qnW + (size_t)row * Dd;
    } else {
        const int r2 = row - Hh * Qn;
        src = keyMem + (size_t)r2 * Dd;
        dst = knW + (size_t)r2 * Dd;
    }

    const float4 v = ((const float4*)src)[lane];
    float ss = v.x * v.x + v.y * v.y + v.z * v.z + v.w * v.w;
#pragma unroll
    for (int off = 16; off >= 1; off >>= 1) ss += __shfl_xor(ss, off, 32);
    const float scale = 1.0f / fmaxf(sqrtf(ss), 1e-11f);

    dst[lane * 4 + 0] = (__bf16)(v.x * scale);
    dst[lane * 4 + 1] = (__bf16)(v.y * scale);
    dst[lane * 4 + 2] = (__bf16)(v.z * scale);
    dst[lane * 4 + 3] = (__bf16)(v.w * scale);
}

// ------- kernel 2: fused bf16-WMMA similarity GEMM + streaming top-16 +
//                   value gather/weighted-sum + sigmoid-gated blend -------
__launch_bounds__(128)
__global__ void praxis_knn_kernel(const __bf16* __restrict__ qnW,
                                  const __bf16* __restrict__ knW,
                                  const float* __restrict__ valueMem,
                                  const float* __restrict__ outputs,
                                  const float* __restrict__ gate,
                                  float* __restrict__ out)
{
    // manual shared-memory layout: topLDS aliases kt (kt is dead after GEMM loop)
    __shared__ __align__(16) unsigned char smem[QT_BYTES + KT_BYTES];
    __bf16 (*qt)[Dd + PAD] = (__bf16(*)[Dd + PAD])smem;
    __bf16 (*kt)[Dd + PAD] = (__bf16(*)[Dd + PAD])(smem + QT_BYTES);
    unsigned long long (*topLDS)[16] = (unsigned long long(*)[16])(smem + QT_BYTES);

    const int h    = blockIdx.y;
    const int qb   = blockIdx.x;
    const int tid  = threadIdx.x;
    const int lane = tid & 31;
    const int wave = tid >> 5;
    const int g    = lane & 15;
    const int half = lane >> 4;

    // stage the 128x128 bf16 query tile (resident for the whole kernel)
    for (int t = tid; t < TQ * (Dd / 8); t += 128) {
        const int r = t >> 4, cc = t & 15;
        const uint4 vv = *(const uint4*)(qnW + ((size_t)(h * Qn + qb * TQ + r)) * Dd + cc * 8);
        *(uint4*)&qt[r][cc * 8] = vv;
    }
    __syncthreads();

    // preload TWO A-tile fragment sets (32 queries per wave) for all K
    // 16-bit A layout: VGPR0-3 = K(+half*8), VGPR4-7 = K+16(+half*8)
    v16bf afrag[2][4];
#pragma unroll
    for (int t = 0; t < 2; ++t) {
        const int qrow = wave * 32 + t * 16 + g;
#pragma unroll
        for (int kk = 0; kk < 4; ++kk) {
            const int c0 = kk * 32 + half * 8;
            v8bf lo = *(const v8bf*)&qt[qrow][c0];
            v8bf hi = *(const v8bf*)&qt[qrow][c0 + 16];
            afrag[t][kk] = __builtin_shufflevector(lo, hi,
                               0,1,2,3,4,5,6,7,8,9,10,11,12,13,14,15);
        }
    }

    // per-wave top-16 state: query = wave*32 + t*16 + v + half*8, slot = g
    unsigned long long topP[2][8];
    float topMin[2][8];
#pragma unroll
    for (int t = 0; t < 2; ++t)
#pragma unroll
        for (int v = 0; v < 8; ++v) { topP[t][v] = 0ull; topMin[t][v] = -__builtin_inff(); }

#pragma unroll 1
    for (int cb = 0; cb < Mm; cb += NC) {
        __syncthreads();
        for (int t = tid; t < NC * (Dd / 8); t += 128) {
            const int r = t >> 4, cc = t & 15;
            const uint4 vv = *(const uint4*)(knW + ((size_t)(h * Mm + cb + r)) * Dd + cc * 8);
            *(uint4*)&kt[r][cc * 8] = vv;
        }
        if (cb + NC < Mm)  // gfx1250 global_prefetch_b8 of next chunk
            __builtin_prefetch(knW + ((size_t)(h * Mm + cb + NC + (tid & (NC - 1)))) * Dd, 0, 1);
        __syncthreads();

#pragma unroll 1
        for (int nt = 0; nt < NC / 16; ++nt) {
            // load ALL four K-step B fragments first (loads overlap WMMAs)
            // 16-bit B layout (K x N): VGPR0-3 = K(+half*8), VGPR4-7 = K+16(+half*8)
            v16bf bf[4];
            const int krow = nt * 16 + g;
#pragma unroll
            for (int kk = 0; kk < 4; ++kk) {
                const int c0 = kk * 32 + half * 8;
                v8bf lo = *(const v8bf*)&kt[krow][c0];
                v8bf hi = *(const v8bf*)&kt[krow][c0 + 16];
                bf[kk] = __builtin_shufflevector(lo, hi,
                             0,1,2,3,4,5,6,7,8,9,10,11,12,13,14,15);
            }
            // each B fragment feeds TWO WMMAs (two A tiles) -> half the LDS/FLOP
            v8f c0 = {}, c1 = {};
#pragma unroll
            for (int kk = 0; kk < 4; ++kk)
                c0 = __builtin_amdgcn_wmma_f32_16x16x32_bf16(
                         false, afrag[0][kk], false, bf[kk], (short)0, c0, false, false);
#pragma unroll
            for (int kk = 0; kk < 4; ++kk)
                c1 = __builtin_amdgcn_wmma_f32_16x16x32_bf16(
                         false, afrag[1][kk], false, bf[kk], (short)0, c1, false, false);

            const unsigned idxBase = (unsigned)(cb + nt * 16);
            topk_update(c0, topP[0], topMin[0], idxBase, g);
            topk_update(c1, topP[1], topMin[1], idxBase, g);
        }
    }

    // park per-query top-16 in LDS (aliases kt, which is now dead)
    __syncthreads();
#pragma unroll
    for (int t = 0; t < 2; ++t)
#pragma unroll
        for (int v = 0; v < 8; ++v) {
            const int q = wave * 32 + t * 16 + v + half * 8;
            topLDS[q][g] = topP[t][v];
        }
    __syncthreads();

    // weighted sum over gathered value rows + gated blend (raw reshape semantics)
    for (int t = tid; t < TQ * (Dd / 4); t += 128) {
        const int q  = t >> 5;
        const int d0 = (t & 31) * 4;
        float a0 = 0.f, a1 = 0.f, a2 = 0.f, a3 = 0.f;
#pragma unroll
        for (int s = 0; s < 16; ++s) {
            const unsigned long long p = topLDS[q][s];
            const float sc    = unpackScore(p);
            const unsigned mi = (unsigned)(p & 0xffffffffull);
            const float4 vr = *(const float4*)(valueMem + ((size_t)(h * Mm + mi)) * Dd + d0);
            a0 += sc * vr.x; a1 += sc * vr.y; a2 += sc * vr.z; a3 += sc * vr.w;
        }
        const size_t Qg   = (size_t)qb * TQ + q;
        const size_t flat = ((size_t)h * Qn + Qg) * Dd + d0;
        const int h_out   = (int)((flat >> 18) & 15);  // torch .view head index
        const float gg    = 1.0f / (1.0f + __expf(-gate[h_out]));
        const float4 ov   = *(const float4*)(outputs + flat);
        float4 res;
        res.x = gg * a0 + (1.0f - gg) * ov.x;
        res.y = gg * a1 + (1.0f - gg) * ov.y;
        res.z = gg * a2 + (1.0f - gg) * ov.z;
        res.w = gg * a3 + (1.0f - gg) * ov.w;
        *(float4*)(out + flat) = res;
    }
}

extern "C" void kernel_launch(void* const* d_in, const int* in_sizes, int n_in,
                              void* d_out, int out_size, void* d_ws, size_t ws_size,
                              hipStream_t stream) {
    (void)in_sizes; (void)n_in; (void)out_size; (void)ws_size;
    // setup_inputs order: inputs, query, key, value, outputs, gate, key_memories, value_memories
    const float* query   = (const float*)d_in[1];
    const float* outputs = (const float*)d_in[4];
    const float* gate    = (const float*)d_in[5];
    const float* keyMem  = (const float*)d_in[6];
    const float* valMem  = (const float*)d_in[7];
    float* out = (float*)d_out;

    __bf16* qnW = (__bf16*)d_ws;                          // 16 MB
    __bf16* knW = qnW + (size_t)Hh * Qn * Dd;             // +16 MB

    // 131072 rows total, 8 rows (waves) per 256-thread block
    praxis_normalize_kernel<<<(2 * Hh * Qn) / 8, 256, 0, stream>>>(query, keyMem, qnW, knW);

    dim3 grid(Qn / TQ, Hh);  // 32 query tiles x 16 heads
    praxis_knn_kernel<<<grid, 128, 0, stream>>>(qnW, knW, valMem, outputs, gate, out);
}
